// AdditiveAttention_4964982194757
// MI455X (gfx1250) — matrix operand order repaired and checked by reading, not compiled, not run on hardware
//
#include <hip/hip_runtime.h>
#include <hip/hip_bf16.h>

typedef __attribute__((ext_vector_type(16))) _Float16 v16h;
typedef __attribute__((ext_vector_type(8)))  _Float16 v8h;
typedef __attribute__((ext_vector_type(8)))  float    v8f;
typedef __attribute__((ext_vector_type(4)))  float    v4f;

#define L_DIM   2048
#define B_DIM   32
#define TD      1024
#define QD      1024
#define W1_LD   2048      // W1 row length (Qd+Td)
#define KT_CNT  32        // K steps of 32 (TD/32)
#define M_ROWS  64        // rows per workgroup (4 M-subtiles of 16)
#define N_WAVES 16        // 512 threads
#define NT_PW   4         // N-tiles per wave (16*4 = 64 tiles = full N)
#define MT_CNT  4         // M-subtiles per wave
#define Z_SPLIT 32        // L-chunks for the apply pass

__device__ __forceinline__ float fast_tanh(float x) {
#if __has_builtin(__builtin_amdgcn_tanhf)
  return __builtin_amdgcn_tanhf(x);        // gfx1250 v_tanh_f32 (hw trans op)
#else
  float ax = fminf(fabsf(x), 15.0f);
  float e  = __expf(-2.0f * ax);           // v_exp_f32 path
  float t  = (1.0f - e) * __builtin_amdgcn_rcpf(1.0f + e);  // v_rcp_f32
  return copysignf(t, x);
#endif
}

// XOR-swizzled LDS index for the f16 A tile (row-major, stride 1024 halves,
// 16B chunks permuted by row so 16-row fragment reads hit all 64 banks).
__device__ __forceinline__ int aswz(int row, int k) {
  return row * 1024 + ((((k >> 3) ^ (row & 15)) << 3) | (k & 7));
}

// ---------------------------------------------------------------------------
// P1: pack W1t = W1[:, QD:] into f16 WMMA B-fragment order.
// Tile (nt, kt): lane = n%16 + 16*(khalf); lane's 16 halves are contiguous K.
// B[k][n] = W1[n][QD + k]  (B = W1t^T for  S = A @ W1t^T)
// ---------------------------------------------------------------------------
__global__ void pack_w1t_kernel(const float* __restrict__ W1,
                                _Float16* __restrict__ Bpack) {
  int g    = blockIdx.x * blockDim.x + threadIdx.x;   // 0..65535 = nt*kt*lane
  int nt   = g >> 10;
  int kt   = (g >> 5) & 31;
  int lane = g & 31;
  int d    = nt * 16 + (lane & 15);
  int c0   = kt * 32 + ((lane >> 4) << 4);
  const float* src = W1 + (size_t)d * W1_LD + QD + c0;
  _Float16* dst = Bpack + (size_t)g * 16;
#pragma unroll
  for (int h = 0; h < 16; ++h) dst[h] = (_Float16)src[h];
}

// ---------------------------------------------------------------------------
// P2: qc[b][d] = b1[d] + sum_c q[b][c] * W1[d][c]   (tiny: 67 MFLOP)
// ---------------------------------------------------------------------------
__global__ void qc_kernel(const float* __restrict__ q,
                          const float* __restrict__ W1,
                          const float* __restrict__ b1,
                          float* __restrict__ qc) {
  int b = blockIdx.y;
  int d = blockIdx.x * blockDim.x + threadIdx.x;  // 0..1023
  const float* qr = q  + (size_t)b * QD;
  const float* wr = W1 + (size_t)d * W1_LD;
  float acc = b1[d];
  for (int c = 0; c < QD; c += 4) {
    acc += qr[c + 0] * wr[c + 0] + qr[c + 1] * wr[c + 1]
         + qr[c + 2] * wr[c + 2] + qr[c + 3] * wr[c + 3];
  }
  qc[(size_t)b * TD + d] = acc;
}

// ---------------------------------------------------------------------------
// S: scores[m] = sum_n v[n] * tanh(qc[b(m)][n] + sum_k A[m][k] * W1t[n][k])
// One workgroup = 64 rows x full N=1024. 16 waves x 4 N-tiles x 4 M-subtiles:
// each B fragment (L2-resident packed W1t) feeds four WMMAs -> 16 WMMA per
// 8 global b128 loads per kt. qc is preloaded as the WMMA C operand.
// LDS: 128KB A tile (XOR-swizzled f16), reused for the cross-wave reduction.
// ---------------------------------------------------------------------------
__global__ void __launch_bounds__(512)
score_kernel(const float* __restrict__ targets,
             const _Float16* __restrict__ Bpack,
             const float* __restrict__ qc,
             const float* __restrict__ v,
             float* __restrict__ scores) {
  __shared__ char smem[M_ROWS * TD * 2];               // 128 KiB
  _Float16* Alds = (_Float16*)smem;
  float*    wsum = (float*)smem;                       // aliased after K-loop

  const int tid = threadIdx.x;
  const int m0  = blockIdx.x * M_ROWS;

  // Stage A tile (64 rows x 1024) to LDS as f16 (coalesced f32 vec4 reads).
  {
    const v4f* tv = reinterpret_cast<const v4f*>(targets + (size_t)m0 * TD);
    for (int i = tid; i < M_ROWS * (TD / 4); i += 512) {
      int row = i >> 8;            // TD/4 == 256 vec4 per row
      int c4  = i & 255;
      v4f val = tv[row * (TD / 4) + c4];
      _Float16* dst = &Alds[aswz(row, c4 * 4)];
      dst[0] = (_Float16)val.x; dst[1] = (_Float16)val.y;
      dst[2] = (_Float16)val.z; dst[3] = (_Float16)val.w;
    }
  }
  __syncthreads();

  const int wave = tid >> 5;      // 0..15
  const int lane = tid & 31;
  const int hi   = lane >> 4;     // half-wave select
  const int ln   = lane & 15;
  const int ntw  = wave * NT_PW;  // this wave's first N-tile

  // Init accumulators with qc tile: C[m][n] layout = VGPR r -> M=r+8*hi, lane->N.
  // m0 is a multiple of 64, so b(m) = (mt*16 + r + 8*hi) & 31.
  v8f acc[MT_CNT][NT_PW];
#pragma unroll
  for (int mt = 0; mt < MT_CNT; ++mt) {
#pragma unroll
    for (int j = 0; j < NT_PW; ++j) {
      const float* qp = qc + (size_t)((ntw + j) * 16 + ln);
#pragma unroll
      for (int r = 0; r < 8; ++r) {
        acc[mt][j][r] = qp[(size_t)((mt * 16 + r + 8 * hi) & 31) * TD];
      }
    }
  }

  const _Float16* Bw   = Bpack + (size_t)ntw * 32 * 512 + lane * 16;
  const int       aoff = hi * 16;

#pragma unroll 1
  for (int kt = 0; kt < KT_CNT; ++kt) {
    const int k0 = kt * 32;
    v16h afrag[MT_CNT];
#pragma unroll
    for (int mt = 0; mt < MT_CNT; ++mt) {
      const int row = mt * 16 + ln;
      const v8h a_lo = *reinterpret_cast<const v8h*>(&Alds[aswz(row, k0 + aoff)]);
      const v8h a_hi = *reinterpret_cast<const v8h*>(&Alds[aswz(row, k0 + 8 + aoff)]);
#pragma unroll
      for (int i = 0; i < 8; ++i) { afrag[mt][i] = a_lo[i]; afrag[mt][i + 8] = a_hi[i]; }
    }
#pragma unroll
    for (int j = 0; j < NT_PW; ++j) {
      const v16h bfrag =
          *reinterpret_cast<const v16h*>(Bw + (size_t)(j * 32 + kt) * 512);
#pragma unroll
      for (int mt = 0; mt < MT_CNT; ++mt) {
        acc[mt][j] = __builtin_amdgcn_wmma_f32_16x16x32_f16(
            false, afrag[mt], false, bfrag, (short)0, acc[mt][j], false, false);
      }
    }
  }

  // Epilogue: tanh, dot with v, reduce across the 16 N-lanes per half-wave.
  float p[MT_CNT][8];
#pragma unroll
  for (int mt = 0; mt < MT_CNT; ++mt)
#pragma unroll
    for (int r = 0; r < 8; ++r) p[mt][r] = 0.0f;
#pragma unroll
  for (int j = 0; j < NT_PW; ++j) {
    const float vn = v[(ntw + j) * 16 + ln];
#pragma unroll
    for (int mt = 0; mt < MT_CNT; ++mt)
#pragma unroll
      for (int r = 0; r < 8; ++r) p[mt][r] += vn * fast_tanh(acc[mt][j][r]);
  }

  __syncthreads();   // A tile dead: safe to alias LDS as wsum[16][64]
#pragma unroll
  for (int mt = 0; mt < MT_CNT; ++mt) {
#pragma unroll
    for (int r = 0; r < 8; ++r) {
      float s = p[mt][r];
      s += __shfl_xor(s, 1, 32);
      s += __shfl_xor(s, 2, 32);
      s += __shfl_xor(s, 4, 32);
      s += __shfl_xor(s, 8, 32);
      if (ln == 0) wsum[wave * M_ROWS + mt * 16 + r + 8 * hi] = s;
    }
  }
  __syncthreads();
  if (tid < M_ROWS) {
    float s = 0.0f;
#pragma unroll
    for (int w = 0; w < N_WAVES; ++w) s += wsum[w * M_ROWS + tid];  // fixed order
    scores[m0 + tid] = s;
  }
}

// ---------------------------------------------------------------------------
// Softmax over L per b; also writes attn_weights output (B,1,L).
// ---------------------------------------------------------------------------
__global__ void softmax_kernel(const float* __restrict__ scores,
                               float* __restrict__ aw,
                               float* __restrict__ out_w) {
  const int b   = blockIdx.x;
  const int tid = threadIdx.x;
  __shared__ float red[256];

  float mx = -3.4e38f;
  for (int l = tid; l < L_DIM; l += 256) mx = fmaxf(mx, scores[l * B_DIM + b]);
  red[tid] = mx; __syncthreads();
  for (int s = 128; s > 0; s >>= 1) {
    if (tid < s) red[tid] = fmaxf(red[tid], red[tid + s]);
    __syncthreads();
  }
  const float smax = red[0];
  __syncthreads();

  float sum = 0.0f;
  for (int l = tid; l < L_DIM; l += 256) {
    float e = __expf(scores[l * B_DIM + b] - smax);
    aw[l * B_DIM + b] = e;
    sum += e;
  }
  red[tid] = sum; __syncthreads();
  for (int s = 128; s > 0; s >>= 1) {
    if (tid < s) red[tid] += red[tid + s];
    __syncthreads();
  }
  const float inv = 1.0f / red[0];

  for (int l = tid; l < L_DIM; l += 256) {
    float a = aw[l * B_DIM + b] * inv;
    aw[l * B_DIM + b] = a;
    out_w[b * L_DIM + l] = a;
  }
}

// ---------------------------------------------------------------------------
// attn_applied[b][d] = sum_l aw[l,b] * targets[l,b,d]  — bandwidth bound;
// float4 loads, split over L for parallelism, fixed-order deterministic reduce.
// ---------------------------------------------------------------------------
__global__ void apply_partial_kernel(const float* __restrict__ targets,
                                     const float* __restrict__ aw,
                                     float* __restrict__ part) {
  const int d4 = threadIdx.x;                     // vec4 column: d = d4*4
  const int b  = blockIdx.y;
  const int z  = blockIdx.z;                      // Z_SPLIT L-chunks
  const int l0 = z * (L_DIM / Z_SPLIT);
  v4f acc = {0.0f, 0.0f, 0.0f, 0.0f};
#pragma unroll 4
  for (int l = l0; l < l0 + (L_DIM / Z_SPLIT); ++l) {
    const int m = l * B_DIM + b;
    const float w = aw[m];
    const v4f t = *reinterpret_cast<const v4f*>(targets + (size_t)m * TD + d4 * 4);
    acc.x += w * t.x; acc.y += w * t.y; acc.z += w * t.z; acc.w += w * t.w;
  }
  *reinterpret_cast<v4f*>(part + ((size_t)z * B_DIM + b) * TD + d4 * 4) = acc;
}

__global__ void apply_reduce_kernel(const float* __restrict__ part,
                                    float* __restrict__ out) {
  const int i = blockIdx.x * 256 + threadIdx.x;   // vec4 index, 0..8191
  v4f s = {0.0f, 0.0f, 0.0f, 0.0f};
#pragma unroll
  for (int z = 0; z < Z_SPLIT; ++z) {
    const v4f t = *reinterpret_cast<const v4f*>(part + (size_t)z * (B_DIM * TD) + i * 4);
    s.x += t.x; s.y += t.y; s.z += t.z; s.w += t.w;
  }
  *reinterpret_cast<v4f*>(out + (size_t)i * 4) = s;
}

// ---------------------------------------------------------------------------
extern "C" void kernel_launch(void* const* d_in, const int* in_sizes, int n_in,
                              void* d_out, int out_size, void* d_ws, size_t ws_size,
                              hipStream_t stream) {
  (void)in_sizes; (void)n_in; (void)out_size; (void)ws_size;
  const float* q  = (const float*)d_in[0];  // (1,B,Qd)
  const float* tg = (const float*)d_in[1];  // (L,B,Td)
  const float* W1 = (const float*)d_in[2];  // (Td, Qd+Td)
  const float* b1 = (const float*)d_in[3];  // (Td,)
  const float* vv = (const float*)d_in[4];  // (1,Td)
  float* out = (float*)d_out;               // [attn_applied(32768) | attn_weights(65536)]

  char* ws = (char*)d_ws;
  _Float16* Bpack = (_Float16*)ws;                       // 2 MiB packed W1t f16
  size_t off = (size_t)2 * 1024 * 1024;
  float* qc     = (float*)(ws + off); off += (size_t)B_DIM * TD * 4;            // 128 KiB
  float* scores = (float*)(ws + off); off += (size_t)L_DIM * B_DIM * 4;         // 256 KiB
  float* aw     = (float*)(ws + off); off += (size_t)L_DIM * B_DIM * 4;         // 256 KiB
  float* part   = (float*)(ws + off); off += (size_t)Z_SPLIT * B_DIM * TD * 4;  // 4 MiB

  pack_w1t_kernel<<<256, 256, 0, stream>>>(W1, Bpack);
  qc_kernel<<<dim3(4, 32), 256, 0, stream>>>(q, W1, b1, qc);
  score_kernel<<<(L_DIM * B_DIM) / M_ROWS, 512, 0, stream>>>(tg, Bpack, qc, vv, scores);
  softmax_kernel<<<B_DIM, 256, 0, stream>>>(scores, aw, out + B_DIM * TD);
  apply_partial_kernel<<<dim3(1, B_DIM, Z_SPLIT), 256, 0, stream>>>(tg, aw, part);
  apply_reduce_kernel<<<(B_DIM * TD) / 1024, 256, 0, stream>>>(part, out);
}
